// GATReg_61125974557481
// MI455X (gfx1250) — matrix operand order repaired and checked by reading, not compiled, not run on hardware
//
#include <hip/hip_runtime.h>
#include <hip/hip_bf16.h>
#include <cstdint>

// ---------------------------------------------------------------------------
// GATv2 3-layer forward for MI455X (gfx1250, wave32).
//  * Node transforms (x@Wl, x@Wr) via v_wmma_f32_16x16x32_f16 (f32 accum).
//  * Edge phase: wave-per-edge (lane = channel), shfl_xor head reduction,
//    segment-max via monotonic-uint atomicMax, segment-sum via float atomicAdd.
// ---------------------------------------------------------------------------

typedef __attribute__((ext_vector_type(16))) _Float16 v16h;
typedef __attribute__((ext_vector_type(8)))  _Float16 v8h;
typedef __attribute__((ext_vector_type(8)))  float    v8f;

#define HEADS 4
#define CHAN  32
#define HID   128
#define NEG_SLOPE 0.2f

static inline int cdiv(int a, int b) { return (a + b - 1) / b; }

// Order-preserving float <-> uint mapping (for atomicMax-based segment max).
__device__ __forceinline__ unsigned f2mono(float f) {
    unsigned u = __float_as_uint(f);
    return (u & 0x80000000u) ? ~u : (u | 0x80000000u);
}
__device__ __forceinline__ float mono2f(unsigned u) {
    unsigned v = (u & 0x80000000u) ? (u & 0x7FFFFFFFu) : ~u;
    return __uint_as_float(v);
}

// ------------------------------ small utility kernels -----------------------

__global__ void k_fill_u32(unsigned* __restrict__ p, unsigned v, int n) {
    int i = blockIdx.x * 256 + threadIdx.x;
    if (i < n) p[i] = v;
}

__global__ void k_f32_to_f16(const float* __restrict__ src, _Float16* __restrict__ dst, int n) {
    int i = blockIdx.x * 256 + threadIdx.x;
    if (i < n) dst[i] = (_Float16)src[i];
}

// W [K,128] row-major  ->  WT [128,K] f16 (so B fragments are contiguous per lane)
__global__ void k_w_transpose_f16(const float* __restrict__ W, _Float16* __restrict__ WT, int K) {
    int i = blockIdx.x * 256 + threadIdx.x;
    if (i < K * HID) {
        int k = i / HID, n = i % HID;
        WT[(size_t)n * K + k] = (_Float16)W[i];
    }
}

// ------------------------------ WMMA GEMM -----------------------------------
// One wave computes rows [tile*16, tile*16+16) x all 128 output cols for BOTH
// Wl and Wr (A fragment reused across 16 WMMAs per K-step).
// A: X16[M,K] f16 row-major.  B: WT[128,K] f16 (transposed weights).
__global__ __launch_bounds__(32) void k_gatv2_gemm(
    const _Float16* __restrict__ X16,
    const _Float16* __restrict__ WlT,
    const _Float16* __restrict__ WrT,
    float* __restrict__ XL, float* __restrict__ XR,
    int M, int K)
{
    const int lane = threadIdx.x & 31;
    const int tile = blockIdx.x;
    const int mh   = lane & 15;          // row-in-tile (A) / col-in-tile (B)
    const int kbA  = (lane >> 4) * 8;    // unused: the A fragment is also loaded at kbB
    const int kbB  = (lane >> 4) * 16;   // B fragment K base for this half-wave

    int row = tile * 16 + mh;
    int arow = row < M ? row : (M - 1);  // clamp (tail tiles) — stores are guarded

    v8f accL[8] = {};
    v8f accR[8] = {};

    for (int k0 = 0; k0 < K; k0 += 32) {
        const _Float16* ap = X16 + (size_t)arow * K + k0;
        v8h a_lo = *(const v8h*)(ap + kbB);        // K = k0+kbB   .. +7   (elems 0..7)
        v8h a_hi = *(const v8h*)(ap + kbB + 8);    // K = k0+kbB+8 .. +7   (elems 8..15)
        v16h a = __builtin_shufflevector(a_lo, a_hi,
                 0,1,2,3,4,5,6,7,8,9,10,11,12,13,14,15);

        #pragma unroll
        for (int t = 0; t < 8; ++t) {
            const int n = t * 16 + mh;
            v16h bl = *(const v16h*)(WlT + (size_t)n * K + k0 + kbB);
            v16h br = *(const v16h*)(WrT + (size_t)n * K + k0 + kbB);
            accL[t] = __builtin_amdgcn_wmma_f32_16x16x32_f16(
                          false, a, false, bl, (short)0, accL[t], false, false);
            accR[t] = __builtin_amdgcn_wmma_f32_16x16x32_f16(
                          false, a, false, br, (short)0, accR[t], false, false);
        }
    }

    // C/D layout: VGPR r -> M = r + (lane>=16 ? 8 : 0), N = lane & 15
    const int r0 = (lane >> 4) * 8;
    #pragma unroll
    for (int t = 0; t < 8; ++t) {
        const int col = t * 16 + mh;
        #pragma unroll
        for (int r = 0; r < 8; ++r) {
            int orow = tile * 16 + r0 + r;
            if (orow < M) {
                XL[(size_t)orow * HID + col] = accL[t][r];
                XR[(size_t)orow * HID + col] = accR[t][r];
            }
        }
    }
}

// ------------------------------ edge kernels (layers 1,2) -------------------

// wave per edge: e[eh] = sum_c leaky(xl[src]+xr[dst]) * att ; atomicMax m[dst,h]
__global__ __launch_bounds__(256) void k_edge_e_max(
    const int* __restrict__ srcA, const int* __restrict__ dstA,
    const float* __restrict__ XL, const float* __restrict__ XR,
    const float* __restrict__ att,
    float* __restrict__ ebuf, unsigned* __restrict__ mbuf, int E)
{
    int wave = (blockIdx.x * 256 + threadIdx.x) >> 5;
    int lane = threadIdx.x & 31;
    if (wave >= E) return;
    int s = srcA[wave], d = dstA[wave];
    const float* xlp = XL + (size_t)s * HID;
    const float* xrp = XR + (size_t)d * HID;
    float part[HEADS];
    #pragma unroll
    for (int h = 0; h < HEADS; ++h) {
        float z = xlp[h * CHAN + lane] + xrp[h * CHAN + lane];
        z = z > 0.f ? z : NEG_SLOPE * z;
        part[h] = z * att[h * CHAN + lane];
    }
    #pragma unroll
    for (int off = 16; off >= 1; off >>= 1) {
        #pragma unroll
        for (int h = 0; h < HEADS; ++h) part[h] += __shfl_xor(part[h], off, 32);
    }
    if (lane == 0) {
        #pragma unroll
        for (int h = 0; h < HEADS; ++h) {
            ebuf[(size_t)wave * HEADS + h] = part[h];
            atomicMax(&mbuf[(size_t)d * HEADS + h], f2mono(part[h]));
        }
    }
}

// thread per (edge, head): ex = exp(e - m[dst,h]); ebuf <- ex; atomicAdd s[dst,h]
__global__ void k_edge_exp_sum(
    const int* __restrict__ dstA, const unsigned* __restrict__ mbuf,
    float* __restrict__ ebuf, float* __restrict__ sbuf, int EH)
{
    int i = blockIdx.x * 256 + threadIdx.x;
    if (i >= EH) return;
    int e = i >> 2, h = i & 3;
    int d = dstA[e];
    float ex = __expf(ebuf[i] - mono2f(mbuf[(size_t)d * HEADS + h]));
    ebuf[i] = ex;
    atomicAdd(&sbuf[(size_t)d * HEADS + h], ex);
}

// wave per edge: out[dst, h*32+c] += xl[src, h*32+c] * alpha[e,h]
__global__ __launch_bounds__(256) void k_edge_accum(
    const int* __restrict__ srcA, const int* __restrict__ dstA,
    const float* __restrict__ XL, const float* __restrict__ ebuf,
    const float* __restrict__ sbuf, float* __restrict__ out, int E)
{
    int wave = (blockIdx.x * 256 + threadIdx.x) >> 5;
    int lane = threadIdx.x & 31;
    if (wave >= E) return;
    int s = srcA[wave], d = dstA[wave];
    #pragma unroll
    for (int h = 0; h < HEADS; ++h) {
        float alpha = ebuf[(size_t)wave * HEADS + h] /
                      (sbuf[(size_t)d * HEADS + h] + 1e-16f);
        atomicAdd(&out[(size_t)d * HID + h * CHAN + lane],
                  XL[(size_t)s * HID + h * CHAN + lane] * alpha);
    }
}

__global__ void k_bias_relu(float* __restrict__ H, const float* __restrict__ b, int n) {
    int i = blockIdx.x * 256 + threadIdx.x;
    if (i < n) {
        float v = H[i] + b[i & (HID - 1)];
        H[i] = v > 0.f ? v : 0.f;
    }
}

// ------------------------------ layer 3 (HID -> 1, single head) -------------

// wave per node: xl3[n] = h[n,:] . Wl3 ; xr3[n] = h[n,:] . Wr3
__global__ __launch_bounds__(256) void k_dot3(
    const float* __restrict__ H, const float* __restrict__ Wl3,
    const float* __restrict__ Wr3, float* __restrict__ XL3,
    float* __restrict__ XR3, int N)
{
    int wave = (blockIdx.x * 256 + threadIdx.x) >> 5;
    int lane = threadIdx.x & 31;
    if (wave >= N) return;
    const float* hp = H + (size_t)wave * HID;
    float al = 0.f, ar = 0.f;
    #pragma unroll
    for (int j = 0; j < HID / 32; ++j) {
        float hv = hp[j * 32 + lane];
        al += hv * Wl3[j * 32 + lane];
        ar += hv * Wr3[j * 32 + lane];
    }
    #pragma unroll
    for (int off = 16; off >= 1; off >>= 1) {
        al += __shfl_xor(al, off, 32);
        ar += __shfl_xor(ar, off, 32);
    }
    if (lane == 0) { XL3[wave] = al; XR3[wave] = ar; }
}

__global__ void k_e3_max(const int* __restrict__ srcA, const int* __restrict__ dstA,
                         const float* __restrict__ XL3, const float* __restrict__ XR3,
                         const float* __restrict__ att3, float* __restrict__ e3,
                         unsigned* __restrict__ m3, int E)
{
    int i = blockIdx.x * 256 + threadIdx.x;
    if (i >= E) return;
    float z = XL3[srcA[i]] + XR3[dstA[i]];
    z = z > 0.f ? z : NEG_SLOPE * z;
    float eh = z * att3[0];
    e3[i] = eh;
    atomicMax(&m3[dstA[i]], f2mono(eh));
}

__global__ void k_e3_exp(const int* __restrict__ dstA, const unsigned* __restrict__ m3,
                         float* __restrict__ e3, float* __restrict__ s3, int E)
{
    int i = blockIdx.x * 256 + threadIdx.x;
    if (i >= E) return;
    int d = dstA[i];
    float ex = __expf(e3[i] - mono2f(m3[d]));
    e3[i] = ex;
    atomicAdd(&s3[d], ex);
}

__global__ void k_e3_acc(const int* __restrict__ srcA, const int* __restrict__ dstA,
                         const float* __restrict__ XL3, const float* __restrict__ e3,
                         const float* __restrict__ s3, float* __restrict__ acc, int E)
{
    int i = blockIdx.x * 256 + threadIdx.x;
    if (i >= E) return;
    int d = dstA[i];
    float alpha = e3[i] / (s3[d] + 1e-16f);
    atomicAdd(&acc[d], XL3[srcA[i]] * alpha);
}

__global__ void k_final_sigmoid(const float* __restrict__ acc, const float* __restrict__ b3,
                                float* __restrict__ out, int N)
{
    int i = blockIdx.x * 256 + threadIdx.x;
    if (i < N) {
        float v = acc[i] + b3[0];
        out[i] = 1.f / (1.f + __expf(-v));
    }
}

// ------------------------------ host orchestration --------------------------

extern "C" void kernel_launch(void* const* d_in, const int* in_sizes, int n_in,
                              void* d_out, int out_size, void* d_ws, size_t ws_size,
                              hipStream_t stream) {
    const float* x    = (const float*)d_in[0];
    const int*   ei   = (const int*)  d_in[1];
    const float* Wl1  = (const float*)d_in[2];
    const float* Wr1  = (const float*)d_in[3];
    const float* att1 = (const float*)d_in[4];
    const float* b1   = (const float*)d_in[5];
    const float* Wl2  = (const float*)d_in[6];
    const float* Wr2  = (const float*)d_in[7];
    const float* att2 = (const float*)d_in[8];
    const float* b2   = (const float*)d_in[9];
    const float* Wl3  = (const float*)d_in[10];
    const float* Wr3  = (const float*)d_in[11];
    const float* att3 = (const float*)d_in[12];
    const float* b3   = (const float*)d_in[13];

    const int N = in_sizes[0] / 64;     // F0 = 64
    const int E = in_sizes[1] / 2;
    const int* srcA = ei;
    const int* dstA = ei + E;

    // workspace carve-up (256B-aligned slots)
    float* ws = (float*)d_ws;
    size_t off = 0;
    auto alloc = [&](size_t nfloats) -> float* {
        float* p = ws + off;
        off += (nfloats + 63) & ~(size_t)63;
        return p;
    };
    float*     XL   = alloc((size_t)N * HID);
    float*     XR   = alloc((size_t)N * HID);
    float*     HB   = alloc((size_t)N * HID);     // layer output / accumulator
    _Float16*  X16  = (_Float16*)alloc((size_t)N * HID / 2);   // N*128 halfs
    _Float16*  WLT  = (_Float16*)alloc(HID * HID / 2);         // 128*128 halfs
    _Float16*  WRT  = (_Float16*)alloc(HID * HID / 2);
    float*     EB   = alloc((size_t)E * HEADS);
    unsigned*  MB   = (unsigned*)alloc((size_t)N * HEADS);
    float*     SB   = alloc((size_t)N * HEADS);
    float*     XL3  = alloc(N);
    float*     XR3  = alloc(N);
    float*     E3   = alloc(E);
    unsigned*  M3   = (unsigned*)alloc(N);
    float*     S3   = alloc(N);
    float*     A3   = alloc(N);

    const int tiles = cdiv(N, 16);
    const int NH    = N * HID;
    const int Nh4   = N * HEADS;
    const int Eh4   = E * HEADS;

    const float* Wls[2]  = {Wl1, Wl2};
    const float* Wrs[2]  = {Wr1, Wr2};
    const float* atts[2] = {att1, att2};
    const float* bs[2]   = {b1, b2};

    for (int l = 0; l < 2; ++l) {
        const int K = (l == 0) ? 64 : HID;
        const float* Xin = (l == 0) ? x : HB;

        // f16 conversion + weight transpose
        k_f32_to_f16<<<cdiv(N * K, 256), 256, 0, stream>>>(Xin, X16, N * K);
        k_w_transpose_f16<<<cdiv(K * HID, 256), 256, 0, stream>>>(Wls[l], WLT, K);
        k_w_transpose_f16<<<cdiv(K * HID, 256), 256, 0, stream>>>(Wrs[l], WRT, K);

        // WMMA GEMM: xl = Xin@Wl, xr = Xin@Wr
        k_gatv2_gemm<<<tiles, 32, 0, stream>>>(X16, WLT, WRT, XL, XR, N, K);

        // clear accumulators / segment buffers
        k_fill_u32<<<cdiv(NH, 256), 256, 0, stream>>>((unsigned*)HB, 0u, NH);
        k_fill_u32<<<cdiv(Nh4, 256), 256, 0, stream>>>(MB, 0u, Nh4);
        k_fill_u32<<<cdiv(Nh4, 256), 256, 0, stream>>>((unsigned*)SB, 0u, Nh4);

        // edge phase: scores + segment max, exp + segment sum, weighted scatter
        k_edge_e_max<<<cdiv(E, 8), 256, 0, stream>>>(srcA, dstA, XL, XR, atts[l], EB, MB, E);
        k_edge_exp_sum<<<cdiv(Eh4, 256), 256, 0, stream>>>(dstA, MB, EB, SB, Eh4);
        k_edge_accum<<<cdiv(E, 8), 256, 0, stream>>>(srcA, dstA, XL, EB, SB, HB, E);

        k_bias_relu<<<cdiv(NH, 256), 256, 0, stream>>>(HB, bs[l], NH);
    }

    // layer 3: HID -> 1, single head, concat=False, sigmoid
    k_dot3<<<cdiv(N, 8), 256, 0, stream>>>(HB, Wl3, Wr3, XL3, XR3, N);
    k_fill_u32<<<cdiv(N, 256), 256, 0, stream>>>(M3, 0u, N);
    k_fill_u32<<<cdiv(N, 256), 256, 0, stream>>>((unsigned*)S3, 0u, N);
    k_fill_u32<<<cdiv(N, 256), 256, 0, stream>>>((unsigned*)A3, 0u, N);
    k_e3_max<<<cdiv(E, 256), 256, 0, stream>>>(srcA, dstA, XL3, XR3, att3, E3, M3, E);
    k_e3_exp<<<cdiv(E, 256), 256, 0, stream>>>(dstA, M3, E3, S3, E);
    k_e3_acc<<<cdiv(E, 256), 256, 0, stream>>>(srcA, dstA, XL3, E3, S3, A3, E);
    k_final_sigmoid<<<cdiv(N, 256), 256, 0, stream>>>(A3, b3, (float*)d_out, N);
}
